// NCELoss_36532991820324
// MI455X (gfx1250) — matrix operand order repaired and checked by reading, not compiled, not run on hardware
//
#include <hip/hip_runtime.h>

// ---------------------------------------------------------------------------
// NCE loss, fused WMMA GEMM + online logsumexp for gfx1250 (MI455X, wave32)
//   A,B: (16, 2048, 256) f32.  out: scalar f32.
//   Phase 1: L2-normalize rows, split to bf16 hi + bf16 lo residual, store in
//            WMMA-fragment-swizzled layout (ISA 7.12.2, 16-bit A / B layouts).
//   Phase 2: per (batch, 128x128 block): K=256 GEMM via v_wmma_f32_16x16x32_bf16
//            (3 WMMAs per tile-k for hi*hi + hi*lo + lo*hi), then exp(s-10)
//            row/col sums + diag, reduced via LDS ds_add_f32 -> global atomics.
//   Phase 3: loss = (sum(10+log(rowsum)) + sum(10+log(colsum)) - 2*diag)/32768
// ---------------------------------------------------------------------------

typedef __attribute__((ext_vector_type(16))) __bf16 v16bf;
typedef __attribute__((ext_vector_type(8)))  float  v8f;

#define NBATCH 16
#define NROWS  2048
#define NCH    256
#define NTOTROWS (NBATCH * NROWS)      // 32768
#define E_ELEMS ((size_t)NBATCH * NROWS * NCH)  // 8388608
#define SHIFT 10.0f
#define INV_TEMP 10.0f

// ---------------------------------------------------------------------------
// Phase 1: normalize one row per wave (32 lanes x 8 floats = 256 channels),
// emit bf16 hi/lo in fragment-swizzled layout.
//   which==0: A-matrix fragment layout (16x32 bf16 A tile, ISA 7.12.2)
//   which==1: B-matrix fragment layout (32x16 bf16 B tile)
// Fragment storage: frag[(row/16)*8 + kc][lane][j], 512 bf16 per fragment.
// ---------------------------------------------------------------------------
__global__ __launch_bounds__(256) void nrm_pack(const float* __restrict__ X,
                                                __bf16* __restrict__ Hi,
                                                __bf16* __restrict__ Lo,
                                                int which) {
  const int wave = threadIdx.x >> 5;
  const int lane = threadIdx.x & 31;
  const int row  = blockIdx.x * 8 + wave;          // 0 .. 32767

  const float4* xp = (const float4*)(X + (size_t)row * NCH);
  float4 x0 = xp[lane * 2 + 0];
  float4 x1 = xp[lane * 2 + 1];

  float ss = x0.x*x0.x + x0.y*x0.y + x0.z*x0.z + x0.w*x0.w
           + x1.x*x1.x + x1.y*x1.y + x1.z*x1.z + x1.w*x1.w;
  ss += __shfl_xor(ss, 1, 32);
  ss += __shfl_xor(ss, 2, 32);
  ss += __shfl_xor(ss, 4, 32);
  ss += __shfl_xor(ss, 8, 32);
  ss += __shfl_xor(ss, 16, 32);
  const float scale = 1.0f / fmaxf(sqrtf(ss), 1e-12f);

  float v[8] = {x0.x, x0.y, x0.z, x0.w, x1.x, x1.y, x1.z, x1.w};
  union { __bf16 h[8]; uint4 u; } hv, lv;
#pragma unroll
  for (int j = 0; j < 8; ++j) {
    float f  = v[j] * scale;
    __bf16 h = (__bf16)f;
    hv.h[j]  = h;
    lv.h[j]  = (__bf16)(f - (float)h);   // residual -> lo part
  }

  // lane handles k-chunk c = lane: k in [8c, 8c+8); kc = c/4, subchunk s = c%4
  const int kc = lane >> 2;
  const int s  = lane & 3;
  int lane_dst, j0;
  if (which == 0) {  // A layout: lanes<16: K{0..7},{16..23}; lanes>=16: K{8..15},{24..31}
    lane_dst = (row & 15) + ((s & 1) ? 16 : 0);
    j0       = (s & 2) ? 8 : 0;
  } else {           // B layout: lanes<16: K{0..7},{16..23}; lanes>=16: K{8..15},{24..31}
    lane_dst = (row & 15) + ((s & 1) ? 16 : 0);
    j0       = (s & 2) ? 8 : 0;
  }
  const size_t off = (((size_t)(row >> 4)) * 8 + kc) * 512 + lane_dst * 16 + j0;
  *(uint4*)(Hi + off) = hv.u;
  *(uint4*)(Lo + off) = lv.u;
}

// ---------------------------------------------------------------------------
// Zero the reduction accumulators (d_ws is poisoned by the harness).
// ---------------------------------------------------------------------------
__global__ void init_acc(float* rowsum, float* colsum, float* diag) {
  const int i = blockIdx.x * blockDim.x + threadIdx.x;
  if (i < NTOTROWS) { rowsum[i] = 0.0f; colsum[i] = 0.0f; }
  if (i == 0) *diag = 0.0f;
}

// ---------------------------------------------------------------------------
// Phase 2: one workgroup = one 128x128 block of S for one batch.
// 8 waves in a 2(m) x 4(n) grid; each wave owns 64x32 = 4x2 WMMA tiles.
// ---------------------------------------------------------------------------
__global__ __launch_bounds__(256) void sim_nce(const __bf16* __restrict__ Ahi,
                                               const __bf16* __restrict__ Alo,
                                               const __bf16* __restrict__ Bhi,
                                               const __bf16* __restrict__ Blo,
                                               float* __restrict__ rowsum,
                                               float* __restrict__ colsum,
                                               float* __restrict__ diagsum) {
  __shared__ float s_row[128];
  __shared__ float s_col[128];
  __shared__ float s_diag;

  const int tid  = threadIdx.x;
  const int lane = tid & 31;
  const int wave = tid >> 5;
  const int wm   = wave >> 2;      // 0..1
  const int wn   = wave & 3;       // 0..3
  const int bx   = blockIdx.x;     // m-block (16)
  const int by   = blockIdx.y;     // n-block (16)
  const int bz   = blockIdx.z;     // batch   (16)

  if (tid < 128) { s_row[tid] = 0.0f; s_col[tid] = 0.0f; }
  if (tid == 0) s_diag = 0.0f;
  __syncthreads();

  v8f acc[4][2];
#pragma unroll
  for (int mi = 0; mi < 4; ++mi)
#pragma unroll
    for (int ni = 0; ni < 2; ++ni) acc[mi][ni] = {};

#pragma unroll
  for (int kc = 0; kc < 8; ++kc) {           // K = 8 * 32 = 256
    v16bf bh[2], bl[2];
#pragma unroll
    for (int ni = 0; ni < 2; ++ni) {
      const int nt = by * 8 + wn * 2 + ni;   // n-tile within batch
      const size_t base = (((size_t)bz * 128 + nt) * 8 + kc) * 512;
      bh[ni] = *((const v16bf*)(Bhi + base) + lane);   // 2x global_load_b128
      bl[ni] = *((const v16bf*)(Blo + base) + lane);
    }
#pragma unroll
    for (int mi = 0; mi < 4; ++mi) {
      const int mt = bx * 8 + wm * 4 + mi;   // m-tile within batch
      const size_t ab = (((size_t)bz * 128 + mt) * 8 + kc) * 512;
      v16bf ah = *((const v16bf*)(Ahi + ab) + lane);
      v16bf al = *((const v16bf*)(Alo + ab) + lane);
#pragma unroll
      for (int ni = 0; ni < 2; ++ni) {
        // hi*hi + hi*lo + lo*hi  (~fp32-accurate dot product)
        acc[mi][ni] = __builtin_amdgcn_wmma_f32_16x16x32_bf16(
            false, ah, false, bh[ni], (short)0, acc[mi][ni], false, false);
        acc[mi][ni] = __builtin_amdgcn_wmma_f32_16x16x32_bf16(
            false, ah, false, bl[ni], (short)0, acc[mi][ni], false, false);
        acc[mi][ni] = __builtin_amdgcn_wmma_f32_16x16x32_bf16(
            false, al, false, bh[ni], (short)0, acc[mi][ni], false, false);
      }
    }
  }

  // ---- epilogue: logits -> exp(s - SHIFT); row/col/diag reductions ----
  const int half = lane >> 4;      // C layout: VGPR v -> M = v + 8*half, N = lane&15
  const int nloc = lane & 15;
#pragma unroll
  for (int mi = 0; mi < 4; ++mi) {
#pragma unroll
    for (int ni = 0; ni < 2; ++ni) {
      const int rbase = wm * 64 + mi * 16;
      const int cbase = wn * 32 + ni * 16;
      const int mtg = bx * 8 + wm * 4 + mi;
      const int ntg = by * 8 + wn * 2 + ni;
      const bool dtile = (mtg == ntg);
      float colp = 0.0f, dpart = 0.0f;
#pragma unroll
      for (int v = 0; v < 8; ++v) {
        const float sv = acc[mi][ni][v] * INV_TEMP;   // logit
        const float e  = __expf(sv - SHIFT);
        colp += e;
        float r = e;                                   // row sum over 16 lanes of this half
        r += __shfl_xor(r, 1, 32);
        r += __shfl_xor(r, 2, 32);
        r += __shfl_xor(r, 4, 32);
        r += __shfl_xor(r, 8, 32);
        const int M = half * 8 + v;
        if (nloc == 0) atomicAdd(&s_row[rbase + M], r);    // ds_add_f32
        if (dtile && nloc == M) dpart += sv;               // diagonal logit
      }
      colp += __shfl_xor(colp, 16, 32);                    // combine M 0..7 with 8..15
      if (lane < 16) atomicAdd(&s_col[cbase + nloc], colp);
      if (dtile) {
        dpart += __shfl_xor(dpart, 1, 32);
        dpart += __shfl_xor(dpart, 2, 32);
        dpart += __shfl_xor(dpart, 4, 32);
        dpart += __shfl_xor(dpart, 8, 32);
        dpart += __shfl_xor(dpart, 16, 32);
        if (lane == 0) atomicAdd(&s_diag, dpart);
      }
    }
  }
  __syncthreads();

  if (tid < 128) {
    atomicAdd(&rowsum[(size_t)bz * NROWS + bx * 128 + tid], s_row[tid]);
  } else {
    atomicAdd(&colsum[(size_t)bz * NROWS + by * 128 + (tid - 128)], s_col[tid - 128]);
  }
  if (tid == 0 && bx == by) atomicAdd(diagsum, s_diag);
}

// ---------------------------------------------------------------------------
// Phase 3: loss = (sum_i [SHIFT+log(rowsum_i)] + sum_j [SHIFT+log(colsum_j)]
//                  - 2*diag) / (B*N)
// ---------------------------------------------------------------------------
__global__ __launch_bounds__(1024) void finish(const float* __restrict__ rowsum,
                                               const float* __restrict__ colsum,
                                               const float* __restrict__ diag,
                                               float* __restrict__ out) {
  __shared__ float parts[32];
  const int tid = threadIdx.x, lane = tid & 31, wave = tid >> 5;
  float acc = 0.0f;
  for (int i = tid; i < 2 * NTOTROWS; i += 1024) {
    const float v = (i < NTOTROWS) ? rowsum[i] : colsum[i - NTOTROWS];
    acc += __logf(v) + SHIFT;
  }
  acc += __shfl_xor(acc, 1, 32);
  acc += __shfl_xor(acc, 2, 32);
  acc += __shfl_xor(acc, 4, 32);
  acc += __shfl_xor(acc, 8, 32);
  acc += __shfl_xor(acc, 16, 32);
  if (lane == 0) parts[wave] = acc;
  __syncthreads();
  if (tid < 32) {
    float t = parts[tid];
    t += __shfl_xor(t, 1, 32);
    t += __shfl_xor(t, 2, 32);
    t += __shfl_xor(t, 4, 32);
    t += __shfl_xor(t, 8, 32);
    t += __shfl_xor(t, 16, 32);
    if (tid == 0) out[0] = (t - 2.0f * diag[0]) * (1.0f / (float)NTOTROWS);
  }
}

// ---------------------------------------------------------------------------
extern "C" void kernel_launch(void* const* d_in, const int* in_sizes, int n_in,
                              void* d_out, int out_size, void* d_ws, size_t ws_size,
                              hipStream_t stream) {
  const float* A = (const float*)d_in[0];
  const float* B = (const float*)d_in[1];
  float* out = (float*)d_out;

  // workspace carve-up: 4 * 16.78 MB bf16 packed operands + reductions (~67.4 MB)
  __bf16* Ahi = (__bf16*)d_ws;
  __bf16* Alo = Ahi + E_ELEMS;
  __bf16* Bhi = Alo + E_ELEMS;
  __bf16* Blo = Bhi + E_ELEMS;
  float* rowsum = (float*)(Blo + E_ELEMS);
  float* colsum = rowsum + NTOTROWS;
  float* diag   = colsum + NTOTROWS;

  init_acc<<<dim3(NTOTROWS / 256), dim3(256), 0, stream>>>(rowsum, colsum, diag);
  nrm_pack<<<dim3(NTOTROWS / 8), dim3(256), 0, stream>>>(A, Ahi, Alo, 0);
  nrm_pack<<<dim3(NTOTROWS / 8), dim3(256), 0, stream>>>(B, Bhi, Blo, 1);
  sim_nce<<<dim3(16, 16, 16), dim3(256), 0, stream>>>(Ahi, Alo, Bhi, Blo,
                                                      rowsum, colsum, diag);
  finish<<<dim3(1), dim3(1024), 0, stream>>>(rowsum, colsum, diag, out);
}
